// RNN_18193481466566
// MI455X (gfx1250) — matrix operand order repaired and checked by reading, not compiled, not run on hardware
//
#include <hip/hip_runtime.h>

typedef __attribute__((ext_vector_type(16))) _Float16 v16h;
typedef __attribute__((ext_vector_type(8)))  float    v8f;

#define SEQ_LEN 262144
#define IN_DIM  100
#define HID     40
#define KP1     128   // padded K (input dim) for phase 1
#define KP3     64    // padded K (hidden dim) for phase 3
#define NP3     112   // padded N (output dim) for phase 3

// ---------------------------------------------------------------------------
// Phase 1: A[seq][40] = s[seq][100] @ Wx[40][100]^T   (WMMA f16 -> f32)
// Block = 128 threads (4 waves); each wave owns one 16-row M tile (64 rows/blk)
// ---------------------------------------------------------------------------
__global__ __launch_bounds__(128) void rnn_phase1_xw(
    const float* __restrict__ s, const float* __restrict__ Wx,
    float* __restrict__ A)
{
    __shared__ _Float16 lds_s[64 * KP1];  // 16 KB, 64x128 f16, zero padded
    __shared__ _Float16 lds_w[48 * KP1];  // 12 KB, Wx as [n][k] zero padded

    const int tid  = threadIdx.x;
    const int lane = tid & 31;
    const int wave = tid >> 5;
    const int m0   = blockIdx.x * 64;

    for (int idx = tid; idx < 64 * KP1; idx += 128) {
        int r = idx >> 7, k = idx & (KP1 - 1);
        float v = (k < IN_DIM) ? s[(size_t)(m0 + r) * IN_DIM + k] : 0.0f;
        lds_s[idx] = (_Float16)v;
    }
    for (int idx = tid; idx < 48 * KP1; idx += 128) {
        int n = idx >> 7, k = idx & (KP1 - 1);
        float v = (n < HID && k < IN_DIM) ? Wx[n * IN_DIM + k] : 0.0f;
        lds_w[idx] = (_Float16)v;
    }
    __syncthreads();

    const int h   = lane >> 4;
    const int l15 = lane & 15;
    const _Float16* myrows = lds_s + (wave * 16) * KP1;

    #pragma unroll
    for (int nt = 0; nt < 3; ++nt) {           // N tiles: 0..15,16..31,32..47
        const int n0 = nt * 16;
        v8f acc = {};
        #pragma unroll
        for (int kc = 0; kc < 4; ++kc) {       // K chunks of 32 (128 total)
            const int kbase = kc * 32;
            v16h a, b;
            #pragma unroll
            for (int e = 0; e < 16; ++e) {
                // A 16x32 f16 layout: VGPR v, half j: K = (v&4?16:0)+h*8+(v&3)*2+j
                int vv = e >> 1, jj = e & 1;
                int ka = kbase + ((vv & 4) ? 16 : 0) + h * 8 + (vv & 3) * 2 + jj;
                a[e] = myrows[l15 * KP1 + ka];
                // B 32x16 f16 layout: same K order as A: K = (e&8?16:0)+h*8+(e&7)
                b[e] = lds_w[(n0 + l15) * KP1 + kbase + ((e & 8) ? 16 : 0) + h * 8 + (e & 7)];
            }
            acc = __builtin_amdgcn_wmma_f32_16x16x32_f16(
                      false, a, false, b, (short)0, acc, false, false);
        }
        #pragma unroll
        for (int r = 0; r < 8; ++r) {          // D: M = r + h*8, N = lane&15
            int m = m0 + wave * 16 + r + h * 8;
            int n = n0 + l15;
            if (n < HID) A[(size_t)m * HID + n] = acc[r];
        }
    }
}

// ---------------------------------------------------------------------------
// Phase 2: sequential scan t_i = tanh(A_i + Wh t_{i-1}); writes T in place
// over A. SINGLE WAVE (32 threads): lanes 0..7 own two rows (j and j+32).
// No barriers on the critical path -- single-wave workgroup barrier ops are
// S_NOP and same-wave LDS ops execute in order, so publish/read of t needs
// only s_wait_dscnt. Wh rows pinned in VGPRs; depth-8 global prefetch ring
// keeps the A-stream load latency off the recurrence.
// ---------------------------------------------------------------------------
__global__ __launch_bounds__(32) void rnn_phase2_scan(
    const float* __restrict__ Wh, const float* __restrict__ init,
    float* __restrict__ AT, float* __restrict__ tfinal)
{
    __shared__ float4 tbank[2][12];           // 2 x 48 floats (40 used)
    const int  j    = threadIdx.x;            // 0..31
    const bool dual = (j < 8);                // this lane also owns row j+32

    float wh0[HID], wh1[HID];
    #pragma unroll
    for (int k = 0; k < HID; ++k) {
        wh0[k] = Wh[j * HID + k];
        wh1[k] = dual ? Wh[(j + 32) * HID + k] : 0.0f;
    }

    {   // init t bank 0 (covers floats 0..47, zero pad above HID)
        float* b0 = (float*)&tbank[0][0];
        b0[j] = init[j];                                        // 0..31
        if (j < 16) b0[32 + j] = (32 + j < HID) ? init[32 + j] : 0.0f; // 32..47
    }

    const int D = 8;                          // prefetch depth
    float abuf0[D], abuf1[D];
    #pragma unroll
    for (int d = 0; d < D; ++d) {
        abuf0[d] = AT[d * HID + j];
        abuf1[d] = dual ? AT[d * HID + 32 + j] : 0.0f;
    }

    float tn0 = 0.0f, tn1 = 0.0f;
    for (int ib = 0; ib < SEQ_LEN; ib += D) {
        #pragma unroll
        for (int u = 0; u < D; ++u) {
            const int i   = ib + u;
            const int cur = i & 1;
            const float a0 = abuf0[u];
            const float a1 = abuf1[u];
            const int ip = i + D;
            if (ip < SEQ_LEN) {
                abuf0[u] = AT[ip * HID + j];
                if (dual) abuf1[u] = AT[ip * HID + 32 + j];
            }

            const float4* t4 = (const float4*)&tbank[cur][0];
            float p0 = a0, p1 = 0.f, p2 = 0.f, p3 = 0.f;   // row j chains
            float q0 = a1, q1 = 0.f, q2 = 0.f, q3 = 0.f;   // row j+32 chains
            #pragma unroll
            for (int g = 0; g < 10; ++g) {    // 10 x b128 LDS broadcast loads
                float4 tv = t4[g];
                p0 = fmaf(wh0[g * 4 + 0], tv.x, p0);
                p1 = fmaf(wh0[g * 4 + 1], tv.y, p1);
                p2 = fmaf(wh0[g * 4 + 2], tv.z, p2);
                p3 = fmaf(wh0[g * 4 + 3], tv.w, p3);
                q0 = fmaf(wh1[g * 4 + 0], tv.x, q0);
                q1 = fmaf(wh1[g * 4 + 1], tv.y, q1);
                q2 = fmaf(wh1[g * 4 + 2], tv.z, q2);
                q3 = fmaf(wh1[g * 4 + 3], tv.w, q3);
            }
            tn0 = tanhf((p0 + p1) + (p2 + p3));
            tn1 = tanhf((q0 + q1) + (q2 + q3));

            float* nb = (float*)&tbank[cur ^ 1][0];
            nb[j] = tn0;                       // publish t_{i+1}
            AT[i * HID + j] = tn0;             // T (in place), fire & forget
            if (dual) {
                nb[32 + j] = tn1;
                AT[i * HID + 32 + j] = tn1;
            }
            // same-wave, in-order LDS: no barrier; stores/loads alias so the
            // compiler preserves order and inserts s_wait_dscnt as needed
        }
    }
    tfinal[j] = tn0;
    if (dual) tfinal[32 + j] = tn1;
}

// ---------------------------------------------------------------------------
// Phase 3: ys[i] = softmax(T[i] @ Wy^T)  (WMMA f16 -> f32, then row softmax)
// Block = 64 threads (2 waves), 16 rows/block; wave0: N tiles 0..3, wave1: 4..6
// Softmax: 4 lanes per row (25 logits each) + __shfl_xor reductions,
// then fully coalesced 64-thread output store.
// ---------------------------------------------------------------------------
__global__ __launch_bounds__(64) void rnn_phase3_softmax(
    const float* __restrict__ T, const float* __restrict__ Wy,
    float* __restrict__ Y)
{
    __shared__ _Float16 lds_t[16 * KP3];      // 2 KB
    __shared__ _Float16 lds_w[NP3 * KP3];     // 14 KB, Wy as [n][k] padded
    __shared__ float    lds_l[16 * NP3];      // 7 KB logits -> exp values
    __shared__ float    lds_inv[16];          // per-row 1/sum

    const int tid  = threadIdx.x;
    const int lane = tid & 31;
    const int wave = tid >> 5;
    const int m0   = blockIdx.x * 16;

    for (int idx = tid; idx < 16 * KP3; idx += 64) {
        int r = idx >> 6, k = idx & (KP3 - 1);
        float v = (k < HID) ? T[(size_t)(m0 + r) * HID + k] : 0.0f;
        lds_t[idx] = (_Float16)v;
    }
    for (int idx = tid; idx < NP3 * KP3; idx += 64) {
        int n = idx >> 6, k = idx & (KP3 - 1);
        float v = (n < IN_DIM && k < HID) ? Wy[n * HID + k] : 0.0f;
        lds_w[idx] = (_Float16)v;
    }
    __syncthreads();

    const int h = lane >> 4, l15 = lane & 15;
    const int nt_begin = (wave == 0) ? 0 : 4;
    const int nt_end   = (wave == 0) ? 4 : 7;
    for (int nt = nt_begin; nt < nt_end; ++nt) {
        const int n0 = nt * 16;
        v8f acc = {};
        #pragma unroll
        for (int kc = 0; kc < 2; ++kc) {      // K = 64 in two chunks of 32
            const int kbase = kc * 32;
            v16h a, b;
            #pragma unroll
            for (int e = 0; e < 16; ++e) {
                int vv = e >> 1, jj = e & 1;
                int ka = kbase + ((vv & 4) ? 16 : 0) + h * 8 + (vv & 3) * 2 + jj;
                a[e] = lds_t[l15 * KP3 + ka];
                b[e] = lds_w[(n0 + l15) * KP3 + kbase + ((e & 8) ? 16 : 0) + h * 8 + (e & 7)];
            }
            acc = __builtin_amdgcn_wmma_f32_16x16x32_f16(
                      false, a, false, b, (short)0, acc, false, false);
        }
        #pragma unroll
        for (int r = 0; r < 8; ++r)
            lds_l[(r + h * 8) * NP3 + n0 + l15] = acc[r];
    }
    __syncthreads();

    {   // parallel softmax: 4 lanes per row, 25 logits per lane
        const int row = tid >> 2;             // 0..15 (rows 0-7 wave0, 8-15 wave1)
        const int sub = tid & 3;
        float* lrow = &lds_l[row * NP3];
        const int nb = sub * 25;

        float mx = -3.402823466e38f;
        #pragma unroll 5
        for (int n = 0; n < 25; ++n) mx = fmaxf(mx, lrow[nb + n]);
        mx = fmaxf(mx, __shfl_xor(mx, 1, 32));
        mx = fmaxf(mx, __shfl_xor(mx, 2, 32));

        float sum = 0.0f;
        #pragma unroll 5
        for (int n = 0; n < 25; ++n) {
            float e = __expf(lrow[nb + n] - mx);
            lrow[nb + n] = e;
            sum += e;
        }
        sum += __shfl_xor(sum, 1, 32);
        sum += __shfl_xor(sum, 2, 32);
        if (sub == 0) lds_inv[row] = 1.0f / sum;
    }
    __syncthreads();

    for (int idx = tid; idx < 16 * IN_DIM; idx += 64) {   // coalesced store
        int row = idx / IN_DIM, n = idx - row * IN_DIM;
        Y[(size_t)(m0 + row) * IN_DIM + n] = lds_l[row * NP3 + n] * lds_inv[row];
    }
}

// ---------------------------------------------------------------------------
extern "C" void kernel_launch(void* const* d_in, const int* in_sizes, int n_in,
                              void* d_out, int out_size, void* d_ws, size_t ws_size,
                              hipStream_t stream)
{
    const float* s    = (const float*)d_in[0];
    const float* init = (const float*)d_in[1];
    const float* Wx   = (const float*)d_in[2];
    const float* Wh   = (const float*)d_in[3];
    const float* Wy   = (const float*)d_in[4];

    float* out    = (float*)d_out;
    float* tfinal = out;          // first 40 floats
    float* Y      = out + HID;    // then [SEQ_LEN x 100]
    float* AT     = (float*)d_ws; // SEQ_LEN*HID floats (A, overwritten by T)

    rnn_phase1_xw     <<<SEQ_LEN / 64, 128, 0, stream>>>(s, Wx, AT);
    rnn_phase2_scan   <<<1,            32,  0, stream>>>(Wh, init, AT, tfinal);
    rnn_phase3_softmax<<<SEQ_LEN / 16, 64,  0, stream>>>(AT, Wy, Y);
}
